// GCN_84670985273387
// MI455X (gfx1250) — matrix operand order repaired and checked by reading, not compiled, not run on hardware
//
#include <hip/hip_runtime.h>
#include <hip/hip_bf16.h>

// ---------------------------------------------------------------------------
// Types for CDNA5 WMMA (wave32)
// ---------------------------------------------------------------------------
typedef __attribute__((ext_vector_type(16))) __bf16          v16bf;
typedef __attribute__((ext_vector_type(8)))  float           v8f;
typedef __attribute__((ext_vector_type(8)))  unsigned short  us8;

__device__ __forceinline__ unsigned short f32_to_bf16_bits(float f) {
  unsigned int u = __float_as_uint(f);
  u += 0x7FFFu + ((u >> 16) & 1u);      // round-to-nearest-even
  return (unsigned short)(u >> 16);
}
__device__ __forceinline__ float bf16_bits_to_f32(unsigned short b) {
  return __uint_as_float(((unsigned int)b) << 16);
}

__device__ __forceinline__ v16bf pack_v16bf(us8 lo, us8 hi) {
  union { us8 h[2]; v16bf v; } u;
  u.h[0] = lo; u.h[1] = hi;
  return u.v;
}

// A fragment: lane L covers row (L&15); elems 0..15 -> K=kb+16*half+i.
// Two b128 loads.
__device__ __forceinline__ v16bf load_afrag(const unsigned short* arow, int kb, int half) {
  const us8* p = (const us8*)(arow + kb);
  return pack_v16bf(p[2 * half], p[2 * half + 1]);
}
// B fragment from pre-transposed Bt[Nout,K]: lane L covers col (L&15),
// elems 0..15 -> K = kb + 16*half + i. One 32B (2x b128) load.
__device__ __forceinline__ v16bf load_bfrag(const unsigned short* brow, size_t K, int j, int kb) {
  const us8* p = (const us8*)(brow + (size_t)(j * 16) * K + kb);
  return pack_v16bf(p[0], p[1]);
}

// ---------------------------------------------------------------------------
// GEMM: Cb[MP,Nout](bf16) = A[MP,K](bf16) * Bt[Nout,K](bf16, pre-transposed)
// One wave: 32 rows x 64 cols (2 row tiles x 4 col tiles, 8 accumulators).
// 2x-unrolled ping-pong register sets: no cur/next rotation movs, fragments
// are written only by VMEM loads and read only by WMMA, so no VALU hazard
// NOPs; s_wait_loadcnt before each WMMA group covers only older loads.
// Requires K % 64 == 0 (true here: 512, 256).
// ---------------------------------------------------------------------------
__global__ __launch_bounds__(128) void gemm_bf16_wmma(
    const unsigned short* __restrict__ A,
    const unsigned short* __restrict__ Bt,
    unsigned short* __restrict__ Cb,
    int MP, int K, int Nout)
{
  const int lane = threadIdx.x & 31;
  const int wave = threadIdx.x >> 5;
  const int pairIdx = blockIdx.x * 4 + wave;       // wave-uniform
  const int m0 = pairIdx * 32;
  if (m0 >= MP) return;                            // whole-wave exit: EXEC all-1s for WMMA
  const int n0 = blockIdx.y << 6;

  const int half = lane >> 4;
  const int l16  = lane & 15;

  v8f acc[2][4] = {};

  const unsigned short* arow0 = A + (size_t)(m0 + l16) * K;
  const unsigned short* arow1 = arow0 + (size_t)16 * K;
  const unsigned short* brow  = Bt + (size_t)(n0 + l16) * K + 16 * half;

  // prologue: set A covers kb = 0
  v16bf a0A = load_afrag(arow0, 0, half);
  v16bf a1A = load_afrag(arow1, 0, half);
  v16bf bA[4];
#pragma unroll
  for (int j = 0; j < 4; ++j) bA[j] = load_bfrag(brow, K, j, 0);

  for (int kb = 0; kb < K; kb += 64) {
    // set B covers kb+32 (always valid: kb <= K-64)
    const int k1 = kb + 32;
    v16bf a0B = load_afrag(arow0, k1, half);
    v16bf a1B = load_afrag(arow1, k1, half);
    v16bf bB[4];
#pragma unroll
    for (int j = 0; j < 4; ++j) bB[j] = load_bfrag(brow, K, j, k1);
    __builtin_prefetch(arow0 + kb + 128, 0, 3);    // global_prefetch_b8 (near scope)
    __builtin_prefetch(arow1 + kb + 128, 0, 3);

    // compute on set A (kb)
#pragma unroll
    for (int j = 0; j < 4; ++j) {
      acc[0][j] = __builtin_amdgcn_wmma_f32_16x16x32_bf16(
          false, a0A, false, bA[j], (short)0, acc[0][j], false, false);
      acc[1][j] = __builtin_amdgcn_wmma_f32_16x16x32_bf16(
          false, a1A, false, bA[j], (short)0, acc[1][j], false, false);
    }

    // refill set A for kb+64 while set-B WMMAs run
    const int k2 = kb + 64;
    if (k2 < K) {
      a0A = load_afrag(arow0, k2, half);
      a1A = load_afrag(arow1, k2, half);
#pragma unroll
      for (int j = 0; j < 4; ++j) bA[j] = load_bfrag(brow, K, j, k2);
    }

    // compute on set B (kb+32)
#pragma unroll
    for (int j = 0; j < 4; ++j) {
      acc[0][j] = __builtin_amdgcn_wmma_f32_16x16x32_bf16(
          false, a0B, false, bB[j], (short)0, acc[0][j], false, false);
      acc[1][j] = __builtin_amdgcn_wmma_f32_16x16x32_bf16(
          false, a1B, false, bB[j], (short)0, acc[1][j], false, false);
    }
  }

  // C lane L, reg r: row = m0 + 16*tile + r + 8*half, col = n0 + j*16 + (L&15)
#pragma unroll
  for (int t = 0; t < 2; ++t) {
    unsigned short* crow = Cb + (size_t)(m0 + 16 * t + 8 * half) * Nout + n0 + l16;
#pragma unroll
    for (int j = 0; j < 4; ++j) {
#pragma unroll
      for (int r = 0; r < 8; ++r) {
        crow[(size_t)r * Nout + j * 16] = f32_to_bf16_bits(acc[t][j][r]);
      }
    }
  }
}

// ---------------------------------------------------------------------------
// Graph / elementwise kernels
// ---------------------------------------------------------------------------
__global__ void fill_f32(float* __restrict__ p, float v, size_t n) {
  size_t i = (size_t)blockIdx.x * blockDim.x + threadIdx.x;
  if (i < n) p[i] = v;
}

__global__ void deg_count(const int* __restrict__ dst, float* __restrict__ deg, int E) {
  int e = blockIdx.x * blockDim.x + threadIdx.x;
  if (e < E) atomicAdd(&deg[dst[e]], 1.0f);
}

__global__ void rsqrt_inplace(float* __restrict__ p, int n) {
  int i = blockIdx.x * blockDim.x + threadIdx.x;
  if (i < n) p[i] = rsqrtf(p[i]);
}

__global__ void cvt_f32_to_bf16(const float* __restrict__ x,
                                unsigned short* __restrict__ y, size_t n) {
  size_t i = (size_t)blockIdx.x * blockDim.x + threadIdx.x;
  if (i < n) y[i] = f32_to_bf16_bits(x[i]);
}

// W[K,Nout] (f32) -> Wt[Nout,K] (bf16)
__global__ void transpose_to_bf16(const float* __restrict__ W,
                                  unsigned short* __restrict__ Wt, int K, int Nout) {
  int i = blockIdx.x * blockDim.x + threadIdx.x;
  if (i < K * Nout) {
    int k = i / Nout, n = i % Nout;
    Wt[(size_t)n * K + k] = f32_to_bf16_bits(W[i]);
  }
}

// One block per edge, lane per bf16 PAIR of features:
//   out[dst] += h[src] * dinv[src]*dinv[dst]   via global_atomic_pk_add_bf16
// Halves gather bytes and atomic-op count vs f32.
__global__ void edge_agg_pk(const unsigned int* __restrict__ h2,   // [N, Fh] packed bf16x2
                            const int* __restrict__ src, const int* __restrict__ dst,
                            const float* __restrict__ dinv,
                            unsigned int* __restrict__ out, int Fh) {
  int e = blockIdx.x;
  int s = src[e], d = dst[e];
  float c = dinv[s] * dinv[d];
  int f = threadIdx.x;
  unsigned int p = h2[(size_t)s * Fh + f];
  float lo = __uint_as_float(p << 16);
  float hi = __uint_as_float(p & 0xffff0000u);
  __hip_bfloat162 val;
  val.x = __float2bfloat16(lo * c);
  val.y = __float2bfloat16(hi * c);
  unsafeAtomicAdd(reinterpret_cast<__hip_bfloat162*>(&out[(size_t)d * Fh + f]), val);
}

// Fused: agg(bf16) + h*dinv^2 (self-loop) + bias -> f32, plus BN partial stats.
// blockDim = F/2, each block covers 256 rows; thread t owns features 2t, 2t+1.
__global__ void finalize_bn(const unsigned int* __restrict__ aggp,   // [N, F/2]
                            const unsigned int* __restrict__ hb2,    // [MP, F/2] bf16x2
                            const float* __restrict__ dinv, const float* __restrict__ bias,
                            float* __restrict__ out,                 // [N, F] f32
                            float* __restrict__ stats, int M, int F) {
  const int t  = threadIdx.x;
  const int f0 = 2 * t;
  const int Fh = F >> 1;
  int r0 = blockIdx.x * 256;
  int r1 = r0 + 256; if (r1 > M) r1 = M;
  const float bias0 = bias[f0], bias1 = bias[f0 + 1];
  float s0 = 0.f, s1 = 0.f, q0 = 0.f, q1 = 0.f;
  for (int r = r0; r < r1; ++r) {
    unsigned int ap = aggp[(size_t)r * Fh + t];
    unsigned int hp = hb2[(size_t)r * Fh + t];
    float di = dinv[r]; float d2 = di * di;
    float v0 = __uint_as_float(ap << 16)         + __uint_as_float(hp << 16)         * d2 + bias0;
    float v1 = __uint_as_float(ap & 0xffff0000u) + __uint_as_float(hp & 0xffff0000u) * d2 + bias1;
    reinterpret_cast<float2*>(out)[(size_t)r * Fh + t] = make_float2(v0, v1);
    s0 += v0; q0 += v0 * v0; s1 += v1; q1 += v1 * v1;
  }
  atomicAdd(&stats[f0], s0);     atomicAdd(&stats[f0 + 1], s1);
  atomicAdd(&stats[F + f0], q0); atomicAdd(&stats[F + f0 + 1], q1);
}

// fused batchnorm + relu + cast to bf16 (next layer's WMMA A operand)
__global__ void bn_relu_to_bf16(const float* __restrict__ x, const float* __restrict__ stats,
                                const float* __restrict__ gamma, const float* __restrict__ beta,
                                unsigned short* __restrict__ y, size_t total, int F, float invM) {
  size_t i = (size_t)blockIdx.x * blockDim.x + threadIdx.x;
  if (i >= total) return;
  int f = (int)(i % F);
  float mean = stats[f] * invM;
  float var  = stats[F + f] * invM - mean * mean;
  float v = (x[i] - mean) * rsqrtf(var + 1e-5f) * gamma[f] + beta[f];
  y[i] = f32_to_bf16_bits(fmaxf(v, 0.f));
}

// Layer-3 epilogue: combine agg + self-loop + bias, then row log_softmax (F=64).
__global__ void combine_logsoftmax64(const unsigned short* __restrict__ aggb,  // [N,64] bf16
                                     const unsigned short* __restrict__ hb,    // [MP,64] bf16
                                     const float* __restrict__ dinv,
                                     const float* __restrict__ bias,
                                     float* __restrict__ y) {
  __shared__ float red[64];
  int row = blockIdx.x;
  int t = threadIdx.x;
  float di = dinv[row];
  float v = bf16_bits_to_f32(aggb[(size_t)row * 64 + t])
          + bf16_bits_to_f32(hb[(size_t)row * 64 + t]) * di * di
          + bias[t];
  red[t] = v; __syncthreads();
  for (int s = 32; s > 0; s >>= 1) { if (t < s) red[t] = fmaxf(red[t], red[t + s]); __syncthreads(); }
  float mx = red[0]; __syncthreads();
  red[t] = expf(v - mx); __syncthreads();
  for (int s = 32; s > 0; s >>= 1) { if (t < s) red[t] += red[t + s]; __syncthreads(); }
  float lse = logf(red[0]);
  y[(size_t)row * 64 + t] = v - mx - lse;
}

// ---------------------------------------------------------------------------
// Host orchestration
// ---------------------------------------------------------------------------
static inline unsigned gdiv(size_t n, unsigned b) { return (unsigned)((n + b - 1) / b); }

static void launch_gemm(const unsigned short* A, const unsigned short* Bt, unsigned short* Cb,
                        int MP, int K, int Nout, hipStream_t s) {
  int pairs = MP / 32;                       // MP is a multiple of 32
  dim3 grid((pairs + 3) / 4, Nout / 64);
  gemm_bf16_wmma<<<grid, dim3(128), 0, s>>>(A, Bt, Cb, MP, K, Nout);
}

extern "C" void kernel_launch(void* const* d_in, const int* in_sizes, int n_in,
                              void* d_out, int out_size, void* d_ws, size_t ws_size,
                              hipStream_t stream) {
  const int IN = 512, HID = 256, OUT = 64;
  const float* x  = (const float*)d_in[0];
  const int*   ei = (const int*)d_in[1];
  const float* W1 = (const float*)d_in[2];  const float* b1 = (const float*)d_in[3];
  const float* W2 = (const float*)d_in[4];  const float* b2 = (const float*)d_in[5];
  const float* W3 = (const float*)d_in[6];  const float* b3 = (const float*)d_in[7];
  const float* g1 = (const float*)d_in[8];  const float* be1 = (const float*)d_in[9];
  const float* g2 = (const float*)d_in[10]; const float* be2 = (const float*)d_in[11];

  const int N  = in_sizes[0] / IN;
  const int E  = in_sizes[1] / 2;
  const int MP = (N + 31) & ~31;             // pad rows for 32-row wave tiles
  const int* src = ei;
  const int* dst = ei + E;

  // workspace carve-out (256B aligned)
  char* w = (char*)d_ws;
  auto carve = [&](size_t bytes) { char* p = w; w += (bytes + 255) & ~(size_t)255; return p; };
  unsigned short* Xb   = (unsigned short*)carve((size_t)MP * IN * 2);   // x as bf16 (padded)
  unsigned short* Hb   = (unsigned short*)carve((size_t)MP * HID * 2);  // activations in (padded)
  unsigned short* Cb   = (unsigned short*)carve((size_t)MP * HID * 2);  // GEMM out bf16 (padded)
  unsigned int*   Aggp = (unsigned int*)carve((size_t)N * HID * 2);     // bf16x2 aggregation acc
  float*          Abuf = (float*)carve((size_t)N * HID * 4);            // f32 BN input
  float*          dinv = (float*)carve((size_t)N * 4);
  unsigned short* W1t  = (unsigned short*)carve((size_t)IN * HID * 2);
  unsigned short* W2t  = (unsigned short*)carve((size_t)HID * HID * 2);
  unsigned short* W3t  = (unsigned short*)carve((size_t)HID * OUT * 2);
  float*          stats = (float*)carve(2 * HID * 4);

  const float invN = 1.0f / (float)N;
  const int FhH = HID / 2, FhO = OUT / 2;

  // degrees (+1 self loop) -> dinv = rsqrt(deg)
  fill_f32<<<gdiv(N, 256), 256, 0, stream>>>(dinv, 1.0f, (size_t)N);
  deg_count<<<gdiv(E, 256), 256, 0, stream>>>(dst, dinv, E);
  rsqrt_inplace<<<gdiv(N, 256), 256, 0, stream>>>(dinv, N);

  // precision prep
  cvt_f32_to_bf16<<<gdiv((size_t)N * IN, 256), 256, 0, stream>>>(x, Xb, (size_t)N * IN);
  transpose_to_bf16<<<gdiv(IN * HID, 256), 256, 0, stream>>>(W1, W1t, IN, HID);
  transpose_to_bf16<<<gdiv(HID * HID, 256), 256, 0, stream>>>(W2, W2t, HID, HID);
  transpose_to_bf16<<<gdiv(HID * OUT, 256), 256, 0, stream>>>(W3, W3t, HID, OUT);

  // ---------------- Layer 1: GCNConv(512->256) + BN + ReLU ----------------
  launch_gemm(Xb, W1t, Cb, MP, IN, HID, stream);
  fill_f32<<<gdiv((size_t)N * FhH, 256), 256, 0, stream>>>((float*)Aggp, 0.f, (size_t)N * FhH);
  edge_agg_pk<<<E, FhH, 0, stream>>>((const unsigned int*)Cb, src, dst, dinv, Aggp, FhH);
  fill_f32<<<gdiv(2 * HID, 256), 256, 0, stream>>>(stats, 0.f, 2 * HID);
  finalize_bn<<<gdiv(N, 256), FhH, 0, stream>>>(Aggp, (const unsigned int*)Cb, dinv, b1,
                                                Abuf, stats, N, HID);
  bn_relu_to_bf16<<<gdiv((size_t)N * HID, 256), 256, 0, stream>>>(Abuf, stats, g1, be1, Hb,
                                                                  (size_t)N * HID, HID, invN);

  // ---------------- Layer 2: GCNConv(256->256) + BN + ReLU ----------------
  launch_gemm(Hb, W2t, Cb, MP, HID, HID, stream);
  fill_f32<<<gdiv((size_t)N * FhH, 256), 256, 0, stream>>>((float*)Aggp, 0.f, (size_t)N * FhH);
  edge_agg_pk<<<E, FhH, 0, stream>>>((const unsigned int*)Cb, src, dst, dinv, Aggp, FhH);
  fill_f32<<<gdiv(2 * HID, 256), 256, 0, stream>>>(stats, 0.f, 2 * HID);
  finalize_bn<<<gdiv(N, 256), FhH, 0, stream>>>(Aggp, (const unsigned int*)Cb, dinv, b2,
                                                Abuf, stats, N, HID);
  bn_relu_to_bf16<<<gdiv((size_t)N * HID, 256), 256, 0, stream>>>(Abuf, stats, g2, be2, Hb,
                                                                  (size_t)N * HID, HID, invN);

  // ---------------- Layer 3: GCNConv(256->64) + log_softmax ----------------
  launch_gemm(Hb, W3t, Cb, MP, HID, OUT, stream);
  fill_f32<<<gdiv((size_t)N * FhO, 256), 256, 0, stream>>>((float*)Aggp, 0.f, (size_t)N * FhO);
  edge_agg_pk<<<E, FhO, 0, stream>>>((const unsigned int*)Cb, src, dst, dinv, Aggp, FhO);
  combine_logsoftmax64<<<N, OUT, 0, stream>>>((const unsigned short*)Aggp, Cb, dinv, b3,
                                              (float*)d_out);
}
